// IDEC_51410758533630
// MI455X (gfx1250) — matrix operand order repaired and checked by reading, not compiled, not run on hardware
//
#include <hip/hip_runtime.h>
#include <hip/hip_bf16.h>

// ---------------------------------------------------------------------------
// IDEC graph autoencoder for MI455X (gfx1250).
// Edge MLPs on v_wmma_f32_16x16x32_bf16; per-graph scatter-mean in LDS.
// Conv kernel fully templated on <DIN, DOUT, MODE> so all tile loops unroll.
// ---------------------------------------------------------------------------

typedef __attribute__((ext_vector_type(16))) __bf16 v16bf;
typedef __attribute__((ext_vector_type(8)))  float  v8f;

#define NODES_C 50
#define B_C     1024
#define N_C     (B_C * NODES_C)      // 51200
#define KNN_C   16
#define E_C     (N_C * KNN_C)        // 819200
#define EPG_C   (NODES_C * KNN_C)    // 800 edges per graph
#define TPG_C   (EPG_C / 16)         // 50 edge-tiles per graph
#define BN_EPS_C 1e-5f

// ------------------------- small utility kernels ---------------------------

__global__ void k_deg(const int* __restrict__ dst, float* __restrict__ deg, int E) {
    int e = blockIdx.x * blockDim.x + threadIdx.x;
    if (e < E) atomicAdd(&deg[dst[e]], 1.0f);
}

__global__ void k_deginv(const float* __restrict__ deg, float* __restrict__ dinv, int n) {
    int i = blockIdx.x * blockDim.x + threadIdx.x;
    if (i < n) dinv[i] = 1.0f / fmaxf(deg[i], 1.0f);
}

__global__ void k_init_x(const float* __restrict__ x, __bf16* __restrict__ h, int n) {
    int idx = blockIdx.x * blockDim.x + threadIdx.x;
    if (idx >= n * 64) return;
    int i = idx >> 6, f = idx & 63;
    float v = (f < 6) ? x[i * 6 + f] : 0.0f;
    h[idx] = (__bf16)v;
}

__global__ void k_expand(const float* __restrict__ gfeat, __bf16* __restrict__ h, int n) {
    int idx = blockIdx.x * blockDim.x + threadIdx.x;
    if (idx >= n * 64) return;
    int i = idx >> 6, f = idx & 63;
    h[idx] = (__bf16)gfeat[(i / NODES_C) * 64 + f];
}

__global__ void k_copy(const float* __restrict__ src, float* __restrict__ dst, int n) {
    int i = blockIdx.x * blockDim.x + threadIdx.x;
    if (i < n) dst[i] = src[i];
}

// Repack W [2*din, dout] f32 -> bf16 tiles in exact WMMA B-matrix lane layout:
// wpack[c][nt][lane][e] = W[c*32 + (e<8 ? (lane>>4)*8 + e : 16 + (lane>>4)*8 + e-8)][nt*16 + (lane&15)]
__global__ void k_repack_w(const float* __restrict__ W, __bf16* __restrict__ wpack,
                           int din, int dout) {
    int kChunks = (2 * din + 31) >> 5;
    int ntiles  = (dout + 15) >> 4;
    int tot = kChunks * ntiles * 512;
    int idx = blockIdx.x * blockDim.x + threadIdx.x;
    if (idx >= tot) return;
    int e    = idx & 15;
    int lane = (idx >> 4) & 31;
    int cn   = idx >> 9;
    int nt   = cn % ntiles;
    int c    = cn / ntiles;
    int k    = c * 32 + ((e < 8) ? ((lane >> 4) * 8 + e) : (16 + (lane >> 4) * 8 + (e - 8)));
    int col  = nt * 16 + (lane & 15);
    float v = (k < 2 * din && col < dout) ? W[k * dout + col] : 0.0f;
    wpack[idx] = (__bf16)v;
}

__global__ void k_bn_finalize(const float* __restrict__ sum, const float* __restrict__ sq,
                              const float* __restrict__ gamma, const float* __restrict__ beta,
                              float* __restrict__ scale, float* __restrict__ shift,
                              int dout, float invE) {
    int j = threadIdx.x;
    if (j >= dout) return;
    float mu  = sum[j] * invE;
    float var = sq[j] * invE - mu * mu;
    float inv = rsqrtf(var + BN_EPS_C);
    float sc  = gamma[j] * inv;
    scale[j] = sc;
    shift[j] = beta[j] - mu * sc;
}

// per-graph mean+max pooling over the 32-dim latent node features
__global__ void k_pool(const float* __restrict__ hF, float* __restrict__ g) {
    int warp = (blockIdx.x * blockDim.x + threadIdx.x) >> 5;
    int lane = threadIdx.x & 31;
    if (warp >= B_C) return;
    const float* base = hF + (size_t)warp * NODES_C * 64;
    float s = 0.0f, mx = -3.402823466e38f;
    for (int i = 0; i < NODES_C; ++i) {
        float v = base[i * 64 + lane];
        s += v; mx = fmaxf(mx, v);
    }
    g[warp * 64 + lane]      = s * (1.0f / NODES_C);
    g[warp * 64 + 32 + lane] = mx;
}

__global__ void k_fc_relu(const float* __restrict__ in, const float* __restrict__ W,
                          const float* __restrict__ b, float* __restrict__ out,
                          int rows, int din, int dout) {
    int idx = blockIdx.x * blockDim.x + threadIdx.x;
    if (idx >= rows * dout) return;
    int r = idx / dout, j = idx % dout;
    const float* x = in + (size_t)r * din;
    float a = b[j];
    for (int k = 0; k < din; ++k) a += x[k] * W[k * dout + j];
    out[idx] = fmaxf(a, 0.0f);
}

__global__ void k_cluster_q(const float* __restrict__ z, const float* __restrict__ c,
                            float* __restrict__ q) {
    int b = blockIdx.x * blockDim.x + threadIdx.x;
    if (b >= B_C) return;
    float qs[10]; float tot = 0.0f;
    for (int k = 0; k < 10; ++k) {
        float d2 = 0.0f;
        for (int j = 0; j < 32; ++j) {
            float df = z[b * 32 + j] - c[k * 32 + j];
            d2 += df * df;
        }
        float v = 1.0f / (1.0f + d2);
        qs[k] = v; tot += v;
    }
    float inv = 1.0f / tot;
    for (int k = 0; k < 10; ++k) q[b * 10 + k] = qs[k] * inv;
}

// ----------------------- fused EdgeConv WMMA kernel ------------------------
// One workgroup (256 threads = 8 waves) per graph; graph g owns edges
// [g*800,(g+1)*800) and nodes [g*50,(g+1)*50). Each wave processes 16-edge
// tiles: A = [16 edges x 2*DIN] bf16 built from gathered node rows, B =
// repacked weights staged in LDS. MODE 0: BN stats via LDS reduce + one
// global flush. MODE 1: normalize+relu messages, scatter-mean entirely in
// LDS (ds_add_f32), then write relu(mean) node features straight out.
// All shape loops are compile-time unrolled.
template <int DIN, int DOUT, int MODE>
__global__ void k_edgeconv(const __bf16* __restrict__ hbf,
                           const int* __restrict__ src, const int* __restrict__ dst,
                           const __bf16* __restrict__ wpack,
                           const float* __restrict__ bias,
                           const float* __restrict__ scale, const float* __restrict__ shift,
                           float* __restrict__ statSum, float* __restrict__ statSq,
                           const float* __restrict__ dinv,
                           __bf16* __restrict__ hout, float* __restrict__ fout,
                           int fstride, int fcols) {
    constexpr int KC = (2 * DIN + 31) / 32;      // K chunks of 32
    constexpr int NT = (DOUT + 15) / 16;         // N tiles of 16
    const int g     = blockIdx.x;                // graph id
    const int node0 = g * NODES_C;

    __shared__ __bf16 smW[8192];                 // up to 16 KB repacked weights
    __shared__ float  smAcc[NODES_C * 64];       // 12.8 KB node accumulators / stats
    {
        constexpr int wTotU = (KC * NT * 512) / 2;   // in uints
        const unsigned* wsrc = (const unsigned*)wpack;
        unsigned* wdst = (unsigned*)smW;
        for (int i = threadIdx.x; i < wTotU; i += blockDim.x) wdst[i] = wsrc[i];
    }
    {
        constexpr int zn = (MODE == 0) ? 128 : NODES_C * 64;
        for (int i = threadIdx.x; i < zn; i += blockDim.x) smAcc[i] = 0.0f;
    }
    __syncthreads();

    const int lane = threadIdx.x & 31;
    const int half = lane >> 4;      // K-split half
    const int mrow = lane & 15;      // A row / C-D column within tile
    const int wib  = threadIdx.x >> 5;           // wave in block (0..7)

    // hoist per-lane column constants (col = nt*16 + mrow); statically indexed
    float bcR[NT], scR[NT], shR[NT];
    #pragma unroll
    for (int nt = 0; nt < NT; ++nt) {
        int col = nt * 16 + mrow;
        bool ok = (col < DOUT);
        bcR[nt] = ok ? bias[col] : 0.0f;
        scR[nt] = (MODE != 0 && ok) ? scale[col] : 0.0f;
        shR[nt] = (MODE != 0 && ok) ? shift[col] : 0.0f;
    }

    for (int lt = wib; lt < TPG_C; lt += 8) {    // 50 tiles per graph
        const int e0 = g * EPG_C + (lt << 4);
        const int eM = e0 + mrow;                // this lane's A-matrix row (edge)
        const int si = src[eM], di = dst[eM];
        const __bf16* xi = hbf + (size_t)di * 64;
        const __bf16* xj = hbf + (size_t)si * 64;

        // local destination nodes for the 8 C/D rows this lane produces
        int dl[8];
        if (MODE != 0) {
            #pragma unroll
            for (int r = 0; r < 8; ++r)
                dl[r] = dst[e0 + r + half * 8] - node0;   // 0..49
        }

        union AChunk { v16bf v; __bf16 h[16]; uint4 q[2]; } A[KC];

        if (DIN == 64) {
            // chunks 0,1: pure x_i   (k = 0..63)
            #pragma unroll
            for (int c = 0; c < 2; ++c) {
                int k0 = c * 32 + half * 8;
                A[c].q[0] = *(const uint4*)(xi + k0);
                A[c].q[1] = *(const uint4*)(xi + k0 + 16);
            }
            // chunks 2,3: x_j - x_i (k = 64..127)
            #pragma unroll
            for (int c = 2; c < 4; ++c) {
                int k0 = (c - 2) * 32 + half * 8;
                union AChunk I, J;
                I.q[0] = *(const uint4*)(xi + k0);
                I.q[1] = *(const uint4*)(xi + k0 + 16);
                J.q[0] = *(const uint4*)(xj + k0);
                J.q[1] = *(const uint4*)(xj + k0 + 16);
                #pragma unroll
                for (int e = 0; e < 16; ++e)
                    A[c].h[e] = (__bf16)((float)J.h[e] - (float)I.h[e]);
            }
        } else {
            #pragma unroll
            for (int c = 0; c < KC; ++c) {
                #pragma unroll
                for (int e = 0; e < 16; ++e) {
                    int pos = (e < 8) ? (half * 8 + e) : (16 + half * 8 + (e - 8));
                    int k = c * 32 + pos;
                    float f = 0.0f;
                    if (k < DIN)          f = (float)xi[k];
                    else if (k < 2 * DIN) f = (float)xj[k - DIN] - (float)xi[k - DIN];
                    A[c].h[e] = (__bf16)f;
                }
            }
        }

        #pragma unroll
        for (int nt = 0; nt < NT; ++nt) {
            v8f acc = {0.f, 0.f, 0.f, 0.f, 0.f, 0.f, 0.f, 0.f};
            #pragma unroll
            for (int c = 0; c < KC; ++c) {
                const __bf16* bp = smW + (((c * NT + nt) * 32 + lane) << 4);
                v16bf Bv = *(const v16bf*)bp;
                acc = __builtin_amdgcn_wmma_f32_16x16x32_bf16(
                          false, A[c].v, false, Bv, (short)0, acc, false, false);
            }
            const int  col = nt * 16 + mrow;
            const bool ok  = (col < DOUT);       // constant-true for DOUT%16==0
            const float bc = bcR[nt];

            if (MODE == 0) {
                float s = 0.0f, sq = 0.0f;
                #pragma unroll
                for (int r = 0; r < 8; ++r) {
                    float tv = acc[r] + bc;
                    s += tv; sq += tv * tv;
                }
                s  += __shfl_xor(s, 16, 32);
                sq += __shfl_xor(sq, 16, 32);
                if (lane < 16 && ok) {
                    atomicAdd(&smAcc[col], s);       // LDS reduce
                    atomicAdd(&smAcc[64 + col], sq);
                }
            } else if (ok) {
                const float sc = scR[nt], sh = shR[nt];
                #pragma unroll
                for (int r = 0; r < 8; ++r) {
                    float tv = acc[r] + bc;
                    float mv = fmaxf(tv * sc + sh, 0.0f);
                    atomicAdd(&smAcc[dl[r] * 64 + col], mv);   // LDS scatter
                }
            }
        }
    }
    __syncthreads();

    if (MODE == 0) {
        // single global flush per block
        int j = threadIdx.x;
        if (j < DOUT)                      atomicAdd(&statSum[j], smAcc[j]);
        else if (j >= 64 && j - 64 < DOUT) atomicAdd(&statSq[j - 64], smAcc[j]);
    } else {
        // fused node update: relu(sum/deg) -> bf16 features (+ optional f32 out)
        for (int idx = threadIdx.x; idx < NODES_C * 64; idx += blockDim.x) {
            int nl = idx >> 6, f = idx & 63;
            int node = node0 + nl;
            float v = 0.0f;
            if (f < DOUT) v = fmaxf(smAcc[idx] * dinv[node], 0.0f);
            hout[(size_t)node * 64 + f] = (__bf16)v;
            if (fout && f < fcols) fout[(size_t)node * fstride + f] = v;
        }
    }
}

// --------------------------------- host ------------------------------------

extern "C" void kernel_launch(void* const* d_in, const int* in_sizes, int n_in,
                              void* d_out, int out_size, void* d_ws, size_t ws_size,
                              hipStream_t stream) {
    (void)in_sizes; (void)n_in; (void)out_size; (void)ws_size;
    const int N = N_C, E = E_C, B = B_C;

    const float* x   = (const float*)d_in[0];
    const int*   ei  = (const int*)d_in[1];
    const int*   src = ei;
    const int*   dst = ei + E;
    const float* clu = (const float*)d_in[3];

    struct Conv { const float *W, *b, *beta, *gamma; int din, dout; };
    auto conv_at = [&](int base, int din, int dout) -> Conv {
        return Conv{(const float*)d_in[base], (const float*)d_in[base + 1],
                    (const float*)d_in[base + 2], (const float*)d_in[base + 3], din, dout};
    };
    // params flattened: cluster, dec_convs[0..3]{W,b,beta,gamma}, dec_fc1{W,b},
    // dec_fc2{W,b}, enc_convs[0..3]{...}, enc_fc1{W,b}, enc_fc2{W,b}
    Conv dec0 = conv_at(4, 64, 64), dec1 = conv_at(8, 64, 64);
    Conv dec2 = conv_at(12, 64, 64), dec3 = conv_at(16, 64, 6);
    const float *dfc1W = (const float*)d_in[20], *dfc1b = (const float*)d_in[21];
    const float *dfc2W = (const float*)d_in[22], *dfc2b = (const float*)d_in[23];
    Conv enc0 = conv_at(24, 6, 64), enc1 = conv_at(28, 64, 64);
    Conv enc2 = conv_at(32, 64, 64), enc3 = conv_at(36, 64, 32);
    const float *efc1W = (const float*)d_in[40], *efc1b = (const float*)d_in[41];
    const float *efc2W = (const float*)d_in[42], *efc2b = (const float*)d_in[43];

    // workspace carve-out (~27 MB)
    char* w = (char*)d_ws; size_t off = 0;
    auto alloc = [&](size_t bytes) -> void* {
        off = (off + 255) & ~(size_t)255;
        void* p = w + off; off += bytes; return p;
    };
    __bf16* hbfA    = (__bf16*)alloc((size_t)N * 64 * 2);
    __bf16* hbfB    = (__bf16*)alloc((size_t)N * 64 * 2);
    float*  hF      = (float*)alloc((size_t)N * 64 * 4);
    float*  deg     = (float*)alloc((size_t)N * 4);
    float*  dinv    = (float*)alloc((size_t)N * 4);
    __bf16* wpack   = (__bf16*)alloc(8192 * 2);
    float*  stats   = (float*)alloc(256 * 4);
    float *statSum = stats, *statSq = stats + 64, *scaleB = stats + 128, *shiftB = stats + 192;
    float*  gbuf  = (float*)alloc((size_t)B * 64 * 4);
    float*  gbuf2 = (float*)alloc((size_t)B * 64 * 4);
    float*  zbuf  = (float*)alloc((size_t)B * 32 * 4);
    float*  dbuf1 = (float*)alloc((size_t)B * 32 * 4);
    float*  dbuf2 = (float*)alloc((size_t)B * 64 * 4);

    float* out_xbar = (float*)d_out;             // [N,6]
    float* out_q    = out_xbar + (size_t)N * 6;  // [B,10]
    float* out_z    = out_q + (size_t)B * 10;    // [B,32]

    const int T = 256;
    auto run_conv = [&](const Conv& cv, const __bf16* hin, __bf16* hout,
                        float* fout, int fstride, int fcols) {
        int kChunks = (2 * cv.din + 31) >> 5;
        int ntiles  = (cv.dout + 15) >> 4;
        int wTot = kChunks * ntiles * 512;
        k_repack_w<<<(wTot + T - 1) / T, T, 0, stream>>>(cv.W, wpack, cv.din, cv.dout);
        hipMemsetAsync(statSum, 0, 128 * sizeof(float), stream);

        #define LAUNCH_PAIR(DIN_, DOUT_)                                                   \
            do {                                                                           \
                k_edgeconv<DIN_, DOUT_, 0><<<B, T, 0, stream>>>(hin, src, dst, wpack,      \
                    cv.b, scaleB, shiftB, statSum, statSq, dinv, hout, fout,               \
                    fstride, fcols);                                                       \
                k_bn_finalize<<<1, 64, 0, stream>>>(statSum, statSq, cv.gamma, cv.beta,    \
                    scaleB, shiftB, DOUT_, 1.0f / (float)E);                               \
                k_edgeconv<DIN_, DOUT_, 1><<<B, T, 0, stream>>>(hin, src, dst, wpack,      \
                    cv.b, scaleB, shiftB, statSum, statSq, dinv, hout, fout,               \
                    fstride, fcols);                                                       \
            } while (0)

        if (cv.din == 6)            LAUNCH_PAIR(6, 64);
        else if (cv.dout == 64)     LAUNCH_PAIR(64, 64);
        else if (cv.dout == 32)     LAUNCH_PAIR(64, 32);
        else                        LAUNCH_PAIR(64, 6);
        #undef LAUNCH_PAIR
    };

    // degree (shared by every conv layer)
    hipMemsetAsync(deg, 0, (size_t)N * sizeof(float), stream);
    k_deg<<<(E + T - 1) / T, T, 0, stream>>>(dst, deg, E);
    k_deginv<<<(N + T - 1) / T, T, 0, stream>>>(deg, dinv, N);

    // ------------- encoder -------------
    k_init_x<<<(N * 64 + T - 1) / T, T, 0, stream>>>(x, hbfA, N);
    run_conv(enc0, hbfA, hbfB, nullptr, 0, 0);
    run_conv(enc1, hbfB, hbfA, nullptr, 0, 0);
    run_conv(enc2, hbfA, hbfB, nullptr, 0, 0);
    run_conv(enc3, hbfB, hbfA, hF, 64, 32);
    k_pool<<<B / 8, T, 0, stream>>>(hF, gbuf);
    k_fc_relu<<<(B * 64 + T - 1) / T, T, 0, stream>>>(gbuf, efc1W, efc1b, gbuf2, B, 64, 64);
    k_fc_relu<<<(B * 32 + T - 1) / T, T, 0, stream>>>(gbuf2, efc2W, efc2b, zbuf, B, 64, 32);
    k_copy<<<(B * 32 + T - 1) / T, T, 0, stream>>>(zbuf, out_z, B * 32);
    k_cluster_q<<<(B + T - 1) / T, T, 0, stream>>>(zbuf, clu, out_q);

    // ------------- decoder -------------
    k_fc_relu<<<(B * 32 + T - 1) / T, T, 0, stream>>>(zbuf, dfc1W, dfc1b, dbuf1, B, 32, 32);
    k_fc_relu<<<(B * 64 + T - 1) / T, T, 0, stream>>>(dbuf1, dfc2W, dfc2b, dbuf2, B, 32, 64);
    k_expand<<<(N * 64 + T - 1) / T, T, 0, stream>>>(dbuf2, hbfA, N);
    run_conv(dec0, hbfA, hbfB, nullptr, 0, 0);
    run_conv(dec1, hbfB, hbfA, nullptr, 0, 0);
    run_conv(dec2, hbfA, hbfB, nullptr, 0, 0);
    run_conv(dec3, hbfB, hbfA, out_xbar, 6, 6);
}
